// MultiHeadAtt_10411000725868
// MI455X (gfx1250) — compile-verified, matrix operand order checked by reading, not run on hardware
//
#include <hip/hip_runtime.h>
#include <hip/hip_bf16.h>
#include <stdint.h>

// ---------------------------------------------------------------------------
// Fused MultiHeadAtt for MI455X (gfx1250, wave32, WMMA).
//
// Algebraic folding:
//   score[b,n,l,q] = sum_h value[b,l,q,h] * qhat[b,q,n,h]        (WMMA, K=h)
//   ctx[b,q,n,h]   = sum_l softmax(score)[n,l] * value[b,l,q,h]  (WMMA, K=l)
//   att = WV @ ctx ; out = LN(leaky(WO @ att + bO) + query_h)
// => value (268 MB) is streamed exactly once, double-buffered through the
//    Tensor Data Mover with s_wait_tensorcnt pipelining.
// ---------------------------------------------------------------------------

typedef _Float16 half_t;
typedef __attribute__((ext_vector_type(16))) _Float16 v16h;
typedef __attribute__((ext_vector_type(8)))  float    v8f;
typedef __attribute__((ext_vector_type(4)))  unsigned int v4u;
typedef __attribute__((ext_vector_type(8)))  int      v8i;
typedef __attribute__((ext_vector_type(4)))  int      v4i;

#define B_      4
#define QL_     512
#define VL_     512
#define NHID_   128
#define KEYHID_ 64
#define NHEAD_  10
#define HDIM_   10
#define NHD_    100
#define NPAD    16      // heads padded to 16 for WMMA M dimension

#define ROWSTRIDE 68    // 64 DW + 4 pad DW (TDM pad) -> float4-aligned rows
#define SXS       68    // transpose scratch stride, float4-aligned

#if defined(__has_builtin)
#  if __has_builtin(__builtin_amdgcn_tensor_load_to_lds)
#    define HAVE_TDM 1
#  endif
#endif
#ifndef HAVE_TDM
#  define HAVE_TDM 0
#  warning "gfx1250 TDM builtin not available - using fallback tile loader"
#endif

// ---------------------------------------------------------------------------
// Tile loader: value[b, l0:l0+32, q, 0:64] (fp32) -> LDS rowbuf, row stride 68.
// TDM: 2D descriptor, tile 64x32, tensor_dim0_stride = QL*KEYHID, LDS pad of
// 4 DW every 64 DW (pad_interval=5, pad_amount=3) for aligned+spread rows.
// ---------------------------------------------------------------------------
__device__ __forceinline__ void tile_issue(float* rb, const float* gsrc, int lane) {
#if HAVE_TDM
  uint32_t lds_addr = (uint32_t)(uintptr_t)rb;     // low 32 bits = LDS offset
  uint64_t ga = (uint64_t)(uintptr_t)gsrc;
  v4u g0;
  g0[0] = 1u;                                      // count=1 (valid), user mode
  g0[1] = lds_addr;                                // lds_addr [63:32]
  g0[2] = (uint32_t)(ga & 0xffffffffu);            // global_addr [95:64]
  g0[3] = (uint32_t)((ga >> 32) & 0x01ffffffu) | (2u << 30); // addr[56:32] | type=2
  const uint32_t dim0 = KEYHID_;                   // h extent
  const uint32_t dim1 = VL_;
  const uint32_t tile0 = KEYHID_, tile1 = 32, tile2 = 0;
  const uint64_t str0 = (uint64_t)QL_ * KEYHID_;   // row (l) stride in elements
  v8i g1;
  g1[0] = (int)((2u << 16)        // data_size = 4B
              | (1u << 20)        // pad_enable
              | (5u << 22)        // pad_interval = 64 DW
              | (3u << 25));      // pad_amount = 4 DW
  g1[1] = (int)((dim0 & 0xffffu) << 16);
  g1[2] = (int)(((dim0 >> 16) & 0xffffu) | ((dim1 & 0xffffu) << 16));
  g1[3] = (int)(((dim1 >> 16) & 0xffffu) | (tile0 << 16));
  g1[4] = (int)(tile1 | (tile2 << 16));
  g1[5] = (int)(str0 & 0xffffffffu);
  g1[6] = (int)((str0 >> 32) & 0xffffu);           // stride1 = 0 (2D)
  g1[7] = 0;
  v4i z4 = {0, 0, 0, 0};
#  if defined(__clang_major__) && __clang_major__ >= 23
  v8i z8 = {0, 0, 0, 0, 0, 0, 0, 0};
  __builtin_amdgcn_tensor_load_to_lds(g0, g1, z4, z4, z8, 0);
#  else
  __builtin_amdgcn_tensor_load_to_lds(g0, g1, z4, z4, 0);
#  endif
  (void)lane;
#else
  // Fallback: per-wave cooperative load (same layout, stride 68, aligned).
  for (int idx = lane; idx < 32 * 16; idx += 32) {
    int row = idx >> 4, c4 = idx & 15;
    const float4* gp = (const float4*)(gsrc + (size_t)row * (QL_ * KEYHID_) + 4 * c4);
    *(float4*)(rb + row * ROWSTRIDE + 4 * c4) = *gp;
  }
#endif
}

// pending==1: leave one TDM in flight (double buffer); pending==0: drain.
__device__ __forceinline__ void tile_wait(int pending) {
#if HAVE_TDM
  if (pending) __builtin_amdgcn_s_wait_tensorcnt((short)1);
  else         __builtin_amdgcn_s_wait_tensorcnt((short)0);
#endif
  asm volatile("" ::: "memory");
}

__device__ __forceinline__ v8f zero8() {
  v8f z = {0.f, 0.f, 0.f, 0.f, 0.f, 0.f, 0.f, 0.f};
  return z;
}

// ---------------------------------------------------------------------------
// Prep: qhat[b,q,n,h] = (sum_d (WQ.query_h + bQ)[n*10+d] * WK[n*10+d,h])/sqrt(10)
//       sbias[b,q,n]  = (sum_d qfull[n*10+d] * WK_b[n*10+d])/sqrt(10)
// ---------------------------------------------------------------------------
__global__ __launch_bounds__(128) void qhat_prep(
    const float* __restrict__ query_h,
    const float* __restrict__ WQ_w, const float* __restrict__ WQ_b,
    const float* __restrict__ WK_w, const float* __restrict__ WK_b,
    half_t* __restrict__ qhat, float* __restrict__ sbias) {
  const int bq = blockIdx.x;           // b*QL + q
  const int t  = threadIdx.x;
  __shared__ float qh[NHID_];
  __shared__ float qf[NHD_ + 12];
  qh[t] = query_h[(size_t)bq * NHID_ + t];
  __syncthreads();
  if (t < NHD_) {
    float acc = WQ_b[t];
    const float* wr = WQ_w + (size_t)t * NHID_;
#pragma unroll 8
    for (int h = 0; h < NHID_; ++h) acc += wr[h] * qh[h];
    qf[t] = acc;
  }
  __syncthreads();
  const float rs = 0.31622776601683794f;     // 1/sqrt(HDIM)
#pragma unroll
  for (int e = 0; e < 8; ++e) {
    int idx = t * 8 + e;
    int n = idx >> 6, h = idx & 63;
    float acc = 0.f;
    if (n < NHEAD_) {
#pragma unroll
      for (int d = 0; d < HDIM_; ++d)
        acc += qf[n * HDIM_ + d] * WK_w[(size_t)(n * HDIM_ + d) * KEYHID_ + h];
    }
    qhat[(size_t)bq * (NPAD * KEYHID_) + idx] = (half_t)(acc * rs);
  }
  if (t < NPAD) {
    float sb = 0.f;
    if (t < NHEAD_) {
#pragma unroll
      for (int d = 0; d < HDIM_; ++d) sb += qf[t * HDIM_ + d] * WK_b[t * HDIM_ + d];
    }
    sbias[(size_t)bq * NPAD + t] = sb * rs;
  }
}

// ---------------------------------------------------------------------------
// Main fused kernel. Block = 2 waves (64 thr), one q column per wave;
// grid = B*QL/2 = 1024 blocks (2048 independent streams for the TDM).
// Each wave streams VL in 32-row tiles, double-buffered through the TDM,
// with online softmax. WMMA operand layouts follow the CDNA5 ISA 16-bit
// A/B lane-half K-split and the f32 C-layout (M = r + 8*(lane/16)).
// ---------------------------------------------------------------------------
__global__ __launch_bounds__(64) void attn_main(
    const float* __restrict__ value, const uint8_t* __restrict__ mask,
    const float* __restrict__ query_h,
    const half_t* __restrict__ qhat, const float* __restrict__ sbias,
    const float* __restrict__ WV_w, const float* __restrict__ WV_b,
    const float* __restrict__ WO_w, const float* __restrict__ WO_b,
    const float* __restrict__ ln_g, const float* __restrict__ ln_b,
    float* __restrict__ out) {
  __shared__ float rowbuf[2][2][32 * ROWSTRIDE]; // [wave][double buffer]
  __shared__ float scratch[2][16 * SXS];         // score transpose / ctx park
  __shared__ float attbuf[2][112];               // per-wave att vector

  const int wave = threadIdx.x >> 5;
  const int lane = threadIdx.x & 31;
  const int lh   = lane >> 4;                    // lane-half (K-split select)
  const int ln16 = lane & 15;
  const int bq   = blockIdx.x * 2 + wave;        // b*QL + q
  const int b    = bq >> 9;                      // / QL_
  const int q    = bq & (QL_ - 1);

  float* buf0 = rowbuf[wave][0];
  float* buf1 = rowbuf[wave][1];
  float* sx   = scratch[wave];
  float* ab   = attbuf[wave];

  // ---- loop-invariant operands ---------------------------------------
  const half_t* qptr = qhat + (size_t)bq * (NPAD * KEYHID_);
  v16h aq[2];                                    // A = qhat, 16n x 32h chunks
#pragma unroll
  for (int c = 0; c < 2; ++c) {
    const half_t* p0 = qptr + ln16 * KEYHID_ + c * 32 + 8 * lh;
#pragma unroll
    for (int i = 0; i < 8; ++i) { aq[c][i] = p0[i]; aq[c][8 + i] = p0[16 + i]; }
  }
  float sb[8];
  {
    const float* sp = sbias + (size_t)bq * NPAD + 8 * lh;
#pragma unroll
    for (int r = 0; r < 8; ++r) sb[r] = sp[r];
  }

  float m_run = -1e30f, d_run = 0.f;
  v8f ctx[4];
#pragma unroll
  for (int t = 0; t < 4; ++t) ctx[t] = zero8();

  const float* vbase = value + ((size_t)b * VL_ * QL_ + q) * KEYHID_;
  tile_issue(buf0, vbase, lane);                 // TDM: tile 0

  constexpr int NT = VL_ / 32;                   // 16 tiles
  for (int it = 0; it < NT; ++it) {
    float* rb = (it & 1) ? buf1 : buf0;

    // ---- keep the DMA pipe full: issue t+1, prefetch t+2, wait t -----
    if (it + 1 < NT) {
      tile_issue((it & 1) ? buf0 : buf1,
                 vbase + (size_t)((it + 1) * 32) * (QL_ * KEYHID_), lane);
      if (it + 2 < NT)
        __builtin_prefetch(vbase + (size_t)((it + 2) * 32 + lane) * (QL_ * KEYHID_), 0, 1);
      tile_wait(1);                              // s_wait_tensorcnt 1
    } else {
      tile_wait(0);                              // s_wait_tensorcnt 0
    }
    const int l0 = it * 32;

    // ---- build value fragments from LDS (fp32 -> f16) ----------------
    v16h bs[2][2];                               // scores B: [l-subtile][h-chunk]
#pragma unroll
    for (int lt = 0; lt < 2; ++lt)
#pragma unroll
      for (int c = 0; c < 2; ++c) {
        const float4* p = (const float4*)(rb + (16 * lt + ln16) * ROWSTRIDE
                                          + c * 32 + 8 * lh);
        float4 u0 = p[0], u1 = p[1];             // K = base+0..7
        float4 u2 = p[4], u3 = p[5];             // K = base+16..23
        bs[lt][c][0] = (half_t)u0.x;  bs[lt][c][1] = (half_t)u0.y;
        bs[lt][c][2] = (half_t)u0.z;  bs[lt][c][3] = (half_t)u0.w;
        bs[lt][c][4] = (half_t)u1.x;  bs[lt][c][5] = (half_t)u1.y;
        bs[lt][c][6] = (half_t)u1.z;  bs[lt][c][7] = (half_t)u1.w;
        bs[lt][c][8]  = (half_t)u2.x; bs[lt][c][9]  = (half_t)u2.y;
        bs[lt][c][10] = (half_t)u2.z; bs[lt][c][11] = (half_t)u2.w;
        bs[lt][c][12] = (half_t)u3.x; bs[lt][c][13] = (half_t)u3.y;
        bs[lt][c][14] = (half_t)u3.z; bs[lt][c][15] = (half_t)u3.w;
      }
    v16h bv[4];                                  // ctx B: K=l(32) x N=h(16)
#pragma unroll
    for (int ht = 0; ht < 4; ++ht) {
      const int hcol = 16 * ht + ln16;
#pragma unroll
      for (int i = 0; i < 8; ++i) {
        bv[ht][i]     = (half_t)rb[(8 * lh + i) * ROWSTRIDE + hcol];
        bv[ht][8 + i] = (half_t)rb[(16 + 8 * lh + i) * ROWSTRIDE + hcol];
      }
    }

    // ---- scores: C[n,l] += qhat * value^T ----------------------------
    v8f sc[2];
#pragma unroll
    for (int lt = 0; lt < 2; ++lt) {
      v8f c = zero8();
      c = __builtin_amdgcn_wmma_f32_16x16x32_f16(false, aq[0], false, bs[lt][0],
                                                 (short)0, c, false, false);
      c = __builtin_amdgcn_wmma_f32_16x16x32_f16(false, aq[1], false, bs[lt][1],
                                                 (short)0, c, false, false);
      sc[lt] = c;
    }

    // ---- mask + K-bias (C layout: lane -> l fixed, n = r + 8*lh) -----
#pragma unroll
    for (int lt = 0; lt < 2; ++lt) {
      const int l = l0 + 16 * lt + ln16;
      const bool msk = mask[((size_t)b * VL_ + l) * QL_ + q] != 0;
#pragma unroll
      for (int r = 0; r < 8; ++r)
        sc[lt][r] = msk ? -1e30f : (sc[lt][r] + sb[r]);
    }

    // ---- transpose C-layout -> A-layout via padded LDS ---------------
#pragma unroll
    for (int lt = 0; lt < 2; ++lt)
#pragma unroll
      for (int r = 0; r < 8; ++r)
        sx[(r + 8 * lh) * SXS + 16 * lt + ln16] = sc[lt][r];
    asm volatile("s_wait_dscnt 0" ::: "memory");

    float sa[16];                                // A layout: lane -> head ln16
    {
      const float4* p = (const float4*)(sx + ln16 * SXS + 8 * lh);
      float4 u0 = p[0], u1 = p[1];               // l = base+0..7
      float4 u2 = p[4], u3 = p[5];               // l = base+16..23
      sa[0]=u0.x; sa[1]=u0.y; sa[2]=u0.z; sa[3]=u0.w;
      sa[4]=u1.x; sa[5]=u1.y; sa[6]=u1.z; sa[7]=u1.w;
      sa[8]=u2.x; sa[9]=u2.y; sa[10]=u2.z; sa[11]=u2.w;
      sa[12]=u3.x; sa[13]=u3.y; sa[14]=u3.z; sa[15]=u3.w;
    }

    // ---- online softmax over l (16 regs + lane^16 partner) -----------
    float mt = sa[0];
#pragma unroll
    for (int i = 1; i < 16; ++i) mt = fmaxf(mt, sa[i]);
    mt = fmaxf(mt, __shfl_xor(mt, 16, 32));
    const float mnew = fmaxf(m_run, mt);
    const float corr = __expf(m_run - mnew);
    float psum = 0.f;
#pragma unroll
    for (int i = 0; i < 16; ++i) { sa[i] = __expf(sa[i] - mnew); psum += sa[i]; }
    psum += __shfl_xor(psum, 16, 32);
    d_run = d_run * corr + psum;
    m_run = mnew;

    v16h ap;                                     // p fragment (A, K=l)
#pragma unroll
    for (int i = 0; i < 16; ++i) ap[i] = (half_t)sa[i];

    // ---- rescale ctx accumulators by corr[n] (bpermute broadcast) ----
#pragma unroll
    for (int r = 0; r < 8; ++r) {
      const float cr = __shfl(corr, r + 8 * lh, 32);
#pragma unroll
      for (int t = 0; t < 4; ++t) ctx[t][r] *= cr;
    }

    // ---- ctx: C[n,h] += p * value ------------------------------------
#pragma unroll
    for (int t = 0; t < 4; ++t)
      ctx[t] = __builtin_amdgcn_wmma_f32_16x16x32_f16(false, ap, false, bv[t],
                                                      (short)0, ctx[t], false, false);
  }

  // ---- finalize softmax normalization --------------------------------
#pragma unroll
  for (int r = 0; r < 8; ++r) {
    const float dr = __shfl(d_run, r + 8 * lh, 32);
    const float inv = 1.f / dr;
#pragma unroll
    for (int t = 0; t < 4; ++t) ctx[t][r] *= inv;
  }

  // ---- park ctx[n][h] in LDS for the tiny epilogue -------------------
#pragma unroll
  for (int t = 0; t < 4; ++t)
#pragma unroll
    for (int r = 0; r < 8; ++r)
      sx[(r + 8 * lh) * SXS + 16 * t + ln16] = ctx[t][r];
  asm volatile("s_wait_dscnt 0" ::: "memory");

  // ---- att = WV @ ctx + bV (per-head row of ctx) ---------------------
  for (int oo = lane; oo < 112; oo += 32) {
    float acc = 0.f;
    if (oo < NHD_) {
      acc = WV_b[oo];
      const float* wr = WV_w + (size_t)oo * KEYHID_;
      const float* cr = sx + (oo / HDIM_) * SXS;
#pragma unroll 8
      for (int h = 0; h < KEYHID_; ++h) acc += wr[h] * cr[h];
    }
    ab[oo] = acc;
  }
  asm volatile("s_wait_dscnt 0" ::: "memory");

  // ---- out = LN(leaky(WO@att + bO) + query_h) ------------------------
  float xo[4], ssum = 0.f, ssq = 0.f;
#pragma unroll
  for (int e = 0; e < 4; ++e) {
    const int o = lane + 32 * e;
    float acc = WO_b[o];
    const float* wr = WO_w + (size_t)o * NHD_;
#pragma unroll 4
    for (int j = 0; j < NHD_; ++j) acc += wr[j] * ab[j];
    acc = acc > 0.f ? acc : 0.01f * acc;                   // leaky_relu
    const float x = acc + query_h[(size_t)bq * NHID_ + o];
    xo[e] = x; ssum += x; ssq += x * x;
  }
#pragma unroll
  for (int off = 16; off; off >>= 1) {
    ssum += __shfl_xor(ssum, off, 32);
    ssq  += __shfl_xor(ssq,  off, 32);
  }
  const float mu  = ssum * (1.f / NHID_);
  const float var = ssq * (1.f / NHID_) - mu * mu;         // biased, as torch LN
  const float inv = rsqrtf(var + 1e-5f);
#pragma unroll
  for (int e = 0; e < 4; ++e) {
    const int o = lane + 32 * e;
    out[(size_t)bq * NHID_ + o] = (xo[e] - mu) * inv * ln_g[o] + ln_b[o];
  }
}

// ---------------------------------------------------------------------------
extern "C" void kernel_launch(void* const* d_in, const int* in_sizes, int n_in,
                              void* d_out, int out_size, void* d_ws, size_t ws_size,
                              hipStream_t stream) {
  const float*   query_h = (const float*)d_in[0];
  const float*   value   = (const float*)d_in[1];
  const uint8_t* mask    = (const uint8_t*)d_in[2];
  const float* WQ_w = (const float*)d_in[3];
  const float* WQ_b = (const float*)d_in[4];
  const float* WK_w = (const float*)d_in[5];
  const float* WK_b = (const float*)d_in[6];
  const float* WV_w = (const float*)d_in[7];
  const float* WV_b = (const float*)d_in[8];
  const float* WO_w = (const float*)d_in[9];
  const float* WO_b = (const float*)d_in[10];
  const float* ln_g = (const float*)d_in[11];
  const float* ln_b = (const float*)d_in[12];

  // workspace: qhat (f16, B*QL*16*64 = 4 MiB) + sbias (f32, 128 KiB)
  half_t* qhat  = (half_t*)d_ws;
  float*  sbias = (float*)((char*)d_ws + (size_t)B_ * QL_ * NPAD * KEYHID_ * sizeof(half_t));

  qhat_prep<<<B_ * QL_, 128, 0, stream>>>(query_h, WQ_w, WQ_b, WK_w, WK_b, qhat, sbias);
  attn_main<<<B_ * QL_ / 2, 64, 0, stream>>>(value, mask, query_h, qhat, sbias,
                                             WV_w, WV_b, WO_w, WO_b, ln_g, ln_b,
                                             (float*)d_out);
  (void)in_sizes; (void)n_in; (void)out_size; (void)ws_size;
}
